// FPModule_47090021433722
// MI455X (gfx1250) — compile-verified, matrix operand order checked by reading, not run on hardware
//
#include <hip/hip_runtime.h>
#include <hip/hip_bf16.h>

// ---------------------------------------------------------------------------
// Problem constants (from the reference)
// ---------------------------------------------------------------------------
#define BG    16          // graphs
#define NS_   1024        // coarse points / graph
#define NT_   8192        // fine points / graph
#define CF    128         // coarse feature dim
#define CSK   64          // skip feature dim
#define KNN   3
#define HID   256         // MLP hidden / out dim
#define DIN   192         // CF + CSK
#define NTOT  (BG * NT_)  // 131072 rows
#define EPSF  1e-16f

typedef __attribute__((ext_vector_type(16))) __bf16 v16bf;
typedef __attribute__((ext_vector_type(8)))  float  v8f;

union Frag16 {            // 16 bf16 = 32 bytes = two uint4 loads
    uint4 u[2];
    v16bf v;
};

__device__ __forceinline__ unsigned int f2bf(float f) {
    union { float f; unsigned int u; } x; x.f = f;
    unsigned int r = x.u + 0x7FFFu + ((x.u >> 16) & 1u);   // RNE
    return r >> 16;
}

// ---------------------------------------------------------------------------
// Kernel 1: convert f32 weight [K, N] (row-major) -> bf16 transposed [N, K]
// ---------------------------------------------------------------------------
__global__ __launch_bounds__(256) void cvt_transpose_kernel(
    const float* __restrict__ w, unsigned short* __restrict__ wt,
    int Kdim, int Ndim) {
    int idx = blockIdx.x * 256 + threadIdx.x;
    if (idx >= Kdim * Ndim) return;
    int k = idx / Ndim;
    int n = idx - k * Ndim;                    // coalesced read over n
    wt[(size_t)n * Kdim + k] = (unsigned short)f2bf(w[idx]);
}

// ---------------------------------------------------------------------------
// Kernel 2: kNN(3) inverse-distance interpolation, fused with building the
// bf16 activation matrix H[NTOT, 192] = [ y(128) | x_skip(64) ].
// One block = 256 target points of one graph.
// ---------------------------------------------------------------------------
__global__ __launch_bounds__(256) void knn_interp_kernel(
    const float* __restrict__ x,        // [BG*NS_, CF]
    const float* __restrict__ pos,      // [BG*NS_, 3]
    const float* __restrict__ x_skip,   // [BG*NT_, CSK]
    const float* __restrict__ pos_skip, // [BG*NT_, 3]
    unsigned short* __restrict__ Hmat)  // [NTOT, DIN] bf16
{
    __shared__ float spos[NS_ * 3];     // 12 KB
    __shared__ int   sidx[256 * 3];
    __shared__ float scw [256 * 3];     // pre-normalized weights

    const int tid   = threadIdx.x;
    const int graph = blockIdx.x >> 5;          // 32 chunks per graph
    const int chunk = blockIdx.x & 31;
    const int tBase = graph * NT_ + chunk * 256;
    const int sBase = graph * NS_;

    // stage source positions of this graph
    for (int i = tid; i < NS_ * 3; i += 256)
        spos[i] = pos[(size_t)sBase * 3 + i];
    __syncthreads();

    // ---- phase 1: per-thread top-3 scan -----------------------------------
    {
        const int t = tBase + tid;
        const float tx = pos_skip[(size_t)t * 3 + 0];
        const float ty = pos_skip[(size_t)t * 3 + 1];
        const float tz = pos_skip[(size_t)t * 3 + 2];

        float d0 = 3.4e38f, d1 = 3.4e38f, d2 = 3.4e38f;
        int   i0 = 0, i1 = 0, i2 = 0;
        #pragma unroll 4
        for (int s = 0; s < NS_; ++s) {
            float dx = tx - spos[s * 3 + 0];
            float dy = ty - spos[s * 3 + 1];
            float dz = tz - spos[s * 3 + 2];
            float d  = dx * dx + dy * dy + dz * dz;
            if (d < d2) {
                if (d < d1) {
                    d2 = d1; i2 = i1;
                    if (d < d0) { d1 = d0; i1 = i0; d0 = d; i0 = s; }
                    else        { d1 = d;  i1 = s; }
                } else { d2 = d; i2 = s; }
            }
        }
        float w0 = 1.0f / fmaxf(d0, EPSF);
        float w1 = 1.0f / fmaxf(d1, EPSF);
        float w2 = 1.0f / fmaxf(d2, EPSF);
        float inv = 1.0f / (w0 + w1 + w2);
        sidx[tid * 3 + 0] = i0;  scw[tid * 3 + 0] = w0 * inv;
        sidx[tid * 3 + 1] = i1;  scw[tid * 3 + 1] = w1 * inv;
        sidx[tid * 3 + 2] = i2;  scw[tid * 3 + 2] = w2 * inv;
    }
    __syncthreads();

    // ---- phase 2: wave-per-point cooperative gather ------------------------
    const int wave = tid >> 5;
    const int lane = tid & 31;
    for (int p = wave; p < 256; p += 8) {
        const int   i0 = sBase + sidx[p * 3 + 0];
        const int   i1 = sBase + sidx[p * 3 + 1];
        const int   i2 = sBase + sidx[p * 3 + 2];
        const float c0 = scw[p * 3 + 0];
        const float c1 = scw[p * 3 + 1];
        const float c2 = scw[p * 3 + 2];
        const size_t t = (size_t)tBase + p;
        unsigned short* hrow = Hmat + t * DIN;

        // interpolated features: 32 lanes x float4 = 128 channels
        const float4 f0 = ((const float4*)(x + (size_t)i0 * CF))[lane];
        const float4 f1 = ((const float4*)(x + (size_t)i1 * CF))[lane];
        const float4 f2 = ((const float4*)(x + (size_t)i2 * CF))[lane];
        float4 y;
        y.x = c0 * f0.x + c1 * f1.x + c2 * f2.x;
        y.y = c0 * f0.y + c1 * f1.y + c2 * f2.y;
        y.z = c0 * f0.z + c1 * f1.z + c2 * f2.z;
        y.w = c0 * f0.w + c1 * f1.w + c2 * f2.w;
        uint2 o;
        o.x = f2bf(y.x) | (f2bf(y.y) << 16);
        o.y = f2bf(y.z) | (f2bf(y.w) << 16);
        ((uint2*)hrow)[lane] = o;

        // skip features: 32 lanes x float2 = 64 channels
        const float2 s2 = ((const float2*)(x_skip + t * CSK))[lane];
        ((unsigned int*)(hrow + CF))[lane] = f2bf(s2.x) | (f2bf(s2.y) << 16);
    }
}

// ---------------------------------------------------------------------------
// Kernel 3/4: WMMA GEMM  C[M,256] = A[M,KDIM] * B[KDIM,256] + bias
// A row-major bf16 (ld = KDIM); BT = B transposed bf16 [256, KDIM].
// Each wave computes a 32(M) x 64(N) tile: 2 A-frags x 4 B-frags ->
// 8 independent WMMAs per K-step (each B-frag feeds 2 WMMAs).
// FUSE==1 -> ReLU + bf16 store; FUSE==0 -> f32 store.
// ---------------------------------------------------------------------------
template <int KDIM, int FUSE>
__global__ __launch_bounds__(256) void gemm_wmma_kernel(
    const __bf16* __restrict__ A,
    const __bf16* __restrict__ BT,
    const float*  __restrict__ bias,
    void* __restrict__ out)
{
    const int lane = threadIdx.x & 31;
    const int half = lane >> 4;            // K sub-block selector
    const int m    = lane & 15;            // A row / B column within tile

    const int waveId = threadIdx.x >> 5;
    const int tileId = blockIdx.x * 8 + waveId;       // tilesN = 4
    const int tm = tileId >> 2;
    const int tn = tileId & 3;
    const int mBase = tm * 32;
    const int nBase = tn * 64;

    v8f acc[2][4] = {{v8f{}, v8f{}, v8f{}, v8f{}},
                     {v8f{}, v8f{}, v8f{}, v8f{}}};

    const __bf16* ar0 = A + (size_t)(mBase + m) * KDIM + half * 8;
    const __bf16* ar1 = ar0 + (size_t)16 * KDIM;
    const __bf16* bp0 = BT + (size_t)(nBase +  0 + m) * KDIM + half * 8;
    const __bf16* bp1 = BT + (size_t)(nBase + 16 + m) * KDIM + half * 8;
    const __bf16* bp2 = BT + (size_t)(nBase + 32 + m) * KDIM + half * 8;
    const __bf16* bp3 = BT + (size_t)(nBase + 48 + m) * KDIM + half * 8;

    #pragma unroll
    for (int k0 = 0; k0 < KDIM; k0 += 32) {
        Frag16 fa0, fa1, fb0, fb1, fb2, fb3;
        // group all fragment loads for this K-step first
        fa0.u[0] = *(const uint4*)(ar0 + k0);
        fa0.u[1] = *(const uint4*)(ar0 + k0 + 16);
        fa1.u[0] = *(const uint4*)(ar1 + k0);
        fa1.u[1] = *(const uint4*)(ar1 + k0 + 16);
        fb0.u[0] = *(const uint4*)(bp0 + k0);
        fb0.u[1] = *(const uint4*)(bp0 + k0 + 16);
        fb1.u[0] = *(const uint4*)(bp1 + k0);
        fb1.u[1] = *(const uint4*)(bp1 + k0 + 16);
        fb2.u[0] = *(const uint4*)(bp2 + k0);
        fb2.u[1] = *(const uint4*)(bp2 + k0 + 16);
        fb3.u[0] = *(const uint4*)(bp3 + k0);
        fb3.u[1] = *(const uint4*)(bp3 + k0 + 16);

        // 8 independent WMMAs: each B-frag feeds both M sub-tiles
        acc[0][0] = __builtin_amdgcn_wmma_f32_16x16x32_bf16(
                        false, fa0.v, false, fb0.v, (short)0, acc[0][0], false, false);
        acc[1][0] = __builtin_amdgcn_wmma_f32_16x16x32_bf16(
                        false, fa1.v, false, fb0.v, (short)0, acc[1][0], false, false);
        acc[0][1] = __builtin_amdgcn_wmma_f32_16x16x32_bf16(
                        false, fa0.v, false, fb1.v, (short)0, acc[0][1], false, false);
        acc[1][1] = __builtin_amdgcn_wmma_f32_16x16x32_bf16(
                        false, fa1.v, false, fb1.v, (short)0, acc[1][1], false, false);
        acc[0][2] = __builtin_amdgcn_wmma_f32_16x16x32_bf16(
                        false, fa0.v, false, fb2.v, (short)0, acc[0][2], false, false);
        acc[1][2] = __builtin_amdgcn_wmma_f32_16x16x32_bf16(
                        false, fa1.v, false, fb2.v, (short)0, acc[1][2], false, false);
        acc[0][3] = __builtin_amdgcn_wmma_f32_16x16x32_bf16(
                        false, fa0.v, false, fb3.v, (short)0, acc[0][3], false, false);
        acc[1][3] = __builtin_amdgcn_wmma_f32_16x16x32_bf16(
                        false, fa1.v, false, fb3.v, (short)0, acc[1][3], false, false);
    }

    // epilogue: D VGPR v maps to row (v + 8*half), column = lane&15
    #pragma unroll
    for (int t = 0; t < 4; ++t) {
        const int n  = nBase + t * 16 + m;
        const float bn = bias[n];
        #pragma unroll
        for (int r = 0; r < 2; ++r) {
            #pragma unroll
            for (int v = 0; v < 8; ++v) {
                const size_t row = (size_t)mBase + r * 16 + v + 8 * half;
                float val = acc[r][t][v] + bn;
                if (FUSE) {
                    val = fmaxf(val, 0.0f);
                    ((unsigned short*)out)[row * HID + n] = (unsigned short)f2bf(val);
                } else {
                    ((float*)out)[row * HID + n] = val;
                }
            }
        }
    }
}

// ---------------------------------------------------------------------------
// Kernel 5: tail outputs -- pos_skip copy + batch_skip values
// ---------------------------------------------------------------------------
__global__ __launch_bounds__(256) void tail_kernel(
    const float* __restrict__ pos_skip, float* __restrict__ out_tail)
{
    const int i = blockIdx.x * 256 + threadIdx.x;
    const int nPos = NTOT * 3;
    if (i < nPos) {
        out_tail[i] = pos_skip[i];
    } else if (i < nPos + NTOT) {
        const int j = i - nPos;
        out_tail[i] = (float)(j / NT_);           // batch index per fine point
    }
}

// ---------------------------------------------------------------------------
// Host launcher
// ---------------------------------------------------------------------------
extern "C" void kernel_launch(void* const* d_in, const int* in_sizes, int n_in,
                              void* d_out, int out_size, void* d_ws, size_t ws_size,
                              hipStream_t stream) {
    (void)in_sizes; (void)n_in; (void)out_size; (void)ws_size;

    const float* x        = (const float*)d_in[0];   // [BG*NS_, CF]
    const float* pos      = (const float*)d_in[1];   // [BG*NS_, 3]
    //            batch   =               d_in[2];   // unused (uniform layout)
    const float* x_skip   = (const float*)d_in[3];   // [BG*NT_, CSK]
    const float* pos_skip = (const float*)d_in[4];   // [BG*NT_, 3]
    //            batch_sk=               d_in[5];   // unused (uniform layout)
    const float* w1       = (const float*)d_in[6];   // [DIN, HID]
    const float* b1       = (const float*)d_in[7];   // [HID]
    const float* w2       = (const float*)d_in[8];   // [HID, HID]
    const float* b2       = (const float*)d_in[9];   // [HID]

    // workspace layout (bytes)
    char* ws = (char*)d_ws;
    unsigned short* Hmat = (unsigned short*)(ws);                       // 50,331,648
    unsigned short* A2   = (unsigned short*)(ws + 50331648);            // 67,108,864
    unsigned short* w1T  = (unsigned short*)(ws + 50331648 + 67108864); //     98,304
    unsigned short* w2T  = (unsigned short*)(ws + 50331648 + 67108864 + 98304);

    float* out = (float*)d_out;                       // [NTOT, HID]
    float* out_tail = out + (size_t)NTOT * HID;       // pos_skip + batch_skip

    // 1) weight conversion + transpose
    cvt_transpose_kernel<<<(DIN * HID + 255) / 256, 256, 0, stream>>>(
        w1, w1T, DIN, HID);
    cvt_transpose_kernel<<<(HID * HID + 255) / 256, 256, 0, stream>>>(
        w2, w2T, HID, HID);

    // 2) kNN interpolation -> fused bf16 activation matrix H [NTOT, 192]
    knn_interp_kernel<<<BG * (NT_ / 256), 256, 0, stream>>>(
        x, pos, x_skip, pos_skip, Hmat);

    // 3) GEMM1: H[NTOT,192] x w1 + b1, ReLU, bf16 -> A2[NTOT,256]
    //    tiles: (NTOT/32) * (HID/64) = 16384 wave-tiles, 8 waves/block
    gemm_wmma_kernel<DIN, 1><<<(NTOT / 32) * (HID / 64) / 8, 256, 0, stream>>>(
        (const __bf16*)Hmat, (const __bf16*)w1T, b1, (void*)A2);

    // 4) GEMM2: A2[NTOT,256] x w2 + b2 -> out f32 [NTOT,256]
    gemm_wmma_kernel<HID, 0><<<(NTOT / 32) * (HID / 64) / 8, 256, 0, stream>>>(
        (const __bf16*)A2, (const __bf16*)w2T, b2, (void*)out);

    // 5) tail outputs
    tail_kernel<<<(NTOT * 4 + 255) / 256, 256, 0, stream>>>(pos_skip, out_tail);
}
